// PointnetSAModuleFS_48979807043955
// MI455X (gfx1250) — matrix operand order repaired and checked by reading, not compiled, not run on hardware
//
#include <hip/hip_runtime.h>
#include <hip/hip_bf16.h>

// ---------------- problem constants (from reference) ----------------
#define B_N     2
#define NPTS    16384
#define NPOINT  4096
#define NFEAT   16
#define EPS_BN  1e-5f

// ---------------- WMMA fragment types (wave32, gfx1250) ----------------
typedef __attribute__((ext_vector_type(16))) _Float16 v16h;
typedef __attribute__((ext_vector_type(8)))  _Float16 v8h;
typedef __attribute__((ext_vector_type(8)))  float    v8f;

union F16Frag { v16h v; v8h h[2]; };

// A-matrix 16x32 f16 (M=out-chan, K=in-chan): lane holds row (lane&15);
// half-index h -> K = base8 + (h&7) + (h>>3)*16, base8 = (lane<16)?0:8.
// Both 8-f16 chunks are contiguous in a [row][K] layout.
__device__ __forceinline__ v16h load_frag_a(const _Float16* __restrict__ row,
                                            int k0, int base8) {
  F16Frag u;
  u.h[0] = *(const v8h*)(row + k0 + base8);
  u.h[1] = *(const v8h*)(row + k0 + base8 + 16);
  return u.v;
}

// B-matrix 32x16 f16 (K x N=points): lane holds column (lane&15);
// half-index h -> K = base8 + (h&7) + (h>>3)*16, base8 = base16/2, in a
// [point][K] activation layout.
__device__ __forceinline__ v16h load_frag_b(const _Float16* __restrict__ row,
                                            int k0, int base16) {
  F16Frag u;
  u.h[0] = *(const v8h*)(row + k0 + (base16 >> 1));
  u.h[1] = *(const v8h*)(row + k0 + (base16 >> 1) + 16);
  return u.v;
}

// ---------------- BN-folding: w16[o][c] = w*gamma/sqrt(var+eps), bias = beta - a*mean
__global__ void fold_bn_kernel(const float* __restrict__ w,
                               const float* __restrict__ g,
                               const float* __restrict__ be,
                               const float* __restrict__ me,
                               const float* __restrict__ va,
                               int cout, int cin, int cinpad,
                               _Float16* __restrict__ w16,
                               float* __restrict__ bias)
{
  int e = blockIdx.x * blockDim.x + threadIdx.x;
  if (e >= cout * cinpad) return;
  int o = e / cinpad, c = e - o * cinpad;
  float a = g[o] * rsqrtf(va[o] + EPS_BN);
  w16[e] = (c < cin) ? (_Float16)(w[o * cin + c] * a) : (_Float16)0.f;
  if (c == 0) bias[o] = be[o] - a * me[o];
}

// ---------------- FPS: one workgroup per batch; 16 pts/thread in registers
__global__ __launch_bounds__(1024) void fps_kernel(
    const float* __restrict__ xyz, const float* __restrict__ density,
    float* __restrict__ new_xyz, float* __restrict__ new_density)
{
  const int b = blockIdx.x;
  const int t = threadIdx.x;
  constexpr int PPT = NPTS / 1024;   // 16
  float px[PPT], py[PPT], pz[PPT], d[PPT];
  const float* base = xyz + (size_t)b * NPTS * 3;
  #pragma unroll
  for (int i = 0; i < PPT; ++i) {
    int p = i * 1024 + t;
    px[i] = base[p * 3 + 0];
    py[i] = base[p * 3 + 1];
    pz[i] = base[p * 3 + 2];
    d[i]  = 1e10f;
  }
  __shared__ float swv[32];
  __shared__ int   swi[32];
  __shared__ float cur[3];
  if (t == 0) {
    cur[0] = base[0]; cur[1] = base[1]; cur[2] = base[2];
    new_xyz[(size_t)b * NPOINT * 3 + 0] = cur[0];
    new_xyz[(size_t)b * NPOINT * 3 + 1] = cur[1];
    new_xyz[(size_t)b * NPOINT * 3 + 2] = cur[2];
    new_density[b * NPOINT] = density[(size_t)b * NPTS];
  }
  __syncthreads();
  for (int j = 1; j < NPOINT; ++j) {
    float cx = cur[0], cy = cur[1], cz = cur[2];
    float best = -1.f; int bi = 0;
    #pragma unroll
    for (int i = 0; i < PPT; ++i) {
      float dx = px[i] - cx, dy = py[i] - cy, dz = pz[i] - cz;
      float nd = dx * dx + dy * dy + dz * dz;
      d[i] = fminf(d[i], nd);
      if (d[i] > best) { best = d[i]; bi = i * 1024 + t; }   // ascending idx: > keeps first
    }
    // wave32 reduce, tie-break to smaller index (jnp.argmax semantics)
    #pragma unroll
    for (int off = 16; off > 0; off >>= 1) {
      float ov = __shfl_xor(best, off, 32);
      int   oi = __shfl_xor(bi,   off, 32);
      if (ov > best || (ov == best && oi < bi)) { best = ov; bi = oi; }
    }
    if ((t & 31) == 0) { swv[t >> 5] = best; swi[t >> 5] = bi; }
    __syncthreads();
    if (t < 32) {
      best = swv[t]; bi = swi[t];
      #pragma unroll
      for (int off = 16; off > 0; off >>= 1) {
        float ov = __shfl_xor(best, off, 32);
        int   oi = __shfl_xor(bi,   off, 32);
        if (ov > best || (ov == best && oi < bi)) { best = ov; bi = oi; }
      }
      if (t == 0) {
        cur[0] = base[bi * 3 + 0];
        cur[1] = base[bi * 3 + 1];
        cur[2] = base[bi * 3 + 2];
        new_xyz[((size_t)b * NPOINT + j) * 3 + 0] = cur[0];
        new_xyz[((size_t)b * NPOINT + j) * 3 + 1] = cur[1];
        new_xyz[((size_t)b * NPOINT + j) * 3 + 2] = cur[2];
        new_density[b * NPOINT + j] = density[(size_t)b * NPTS + bi];
      }
    }
    __syncthreads();
  }
}

// ---------------- ball query: one wave per center, ordered append + early exit
template<int K>
__global__ __launch_bounds__(256) void ball_query_kernel(
    const float* __restrict__ xyz, const float* __restrict__ new_xyz,
    int* __restrict__ gidx, int* __restrict__ cnt, float r2)
{
  const int lane = threadIdx.x & 31;
  const int gw = blockIdx.x * 8 + (threadIdx.x >> 5);   // center id over B*NPOINT
  const int b = gw >> 12;                                // /NPOINT
  const float cx = new_xyz[(size_t)gw * 3 + 0];
  const float cy = new_xyz[(size_t)gw * 3 + 1];
  const float cz = new_xyz[(size_t)gw * 3 + 2];
  const float* base = xyz + (size_t)b * NPTS * 3;
  int* out = gidx + (size_t)gw * K;
  int found = 0, first = -1;
  for (int p0 = 0; p0 < NPTS && found < K; p0 += 32) {
    int p = p0 + lane;
    float dx = base[p * 3 + 0] - cx;
    float dy = base[p * 3 + 1] - cy;
    float dz = base[p * 3 + 2] - cz;
    bool in = (dx * dx + dy * dy + dz * dz) < r2;
    unsigned mask = (unsigned)__ballot(in);
    if (first < 0 && mask) first = p0 + __builtin_ctz(mask);
    if (in) {
      int slot = found + __popc(mask & ((1u << lane) - 1u));
      if (slot < K) out[slot] = p;
    }
    found += __popc(mask);
  }
  int fill = (first < 0) ? 0 : first;
  int start = (found < K) ? found : K;
  for (int sl = start + lane; sl < K; sl += 32) out[sl] = fill;
  if (lane == 0) cnt[gw] = found;
}

// ---------------- one WMMA MLP layer over an LDS tile: dst = relu(W*src + bias)
template<int CIN, int COUT, int PB>
__device__ __forceinline__ void mlp_layer(const _Float16* __restrict__ src,  // [PB][CIN] LDS
                                          _Float16* __restrict__ dst,        // [PB][COUT] LDS
                                          const _Float16* __restrict__ W,    // [COUT][CIN] global
                                          const float* __restrict__ bias)
{
  const int lane   = threadIdx.x & 31;
  const int wave   = threadIdx.x >> 5;
  const int l15    = lane & 15;
  const int base8  = (lane < 16) ? 0 : 8;
  const int base16 = (lane < 16) ? 0 : 16;
  constexpr int MT = COUT / 16;
  constexpr int NT = PB / 16;
  for (int t = wave; t < MT * NT; t += 8) {
    int mt = t / NT, nt = t - mt * NT;
    v8f acc = {};
    const _Float16* wrow = W   + (size_t)(mt * 16 + l15) * CIN;
    const _Float16* arow = src + (size_t)(nt * 16 + l15) * CIN;
    #pragma unroll
    for (int k0 = 0; k0 < CIN; k0 += 32) {
      v16h af = load_frag_a(wrow, k0, base8);
      v16h bf = load_frag_b(arow, k0, base16);
      acc = __builtin_amdgcn_wmma_f32_16x16x32_f16(false, af, false, bf,
                                                   (short)0, acc, false, false);
    }
    v8h out;
    #pragma unroll
    for (int i = 0; i < 8; ++i) {
      float v = acc[i] + bias[mt * 16 + base8 + i];    // C/D: row = i + base8
      out[i] = (_Float16)(v > 0.f ? v : 0.f);
    }
    *(v8h*)(dst + (size_t)(nt * 16 + l15) * COUT + mt * 16 + base8) = out;
  }
}

// ---------------- fused: gather/group -> MLP x2 (WMMA, LDS-resident) -> masked max-pool
template<int K, int CB, int C1, int C2>
__global__ __launch_bounds__(256) void fused_scale_kernel(
    const float* __restrict__ xyz, const float* __restrict__ feat,
    const float* __restrict__ new_xyz,
    const int* __restrict__ gidx, const int* __restrict__ cnt,
    const _Float16* __restrict__ W1, const float* __restrict__ b1,
    const _Float16* __restrict__ W2, const float* __restrict__ b2,
    _Float16* __restrict__ aggin, int choff)
{
  constexpr int PB = CB * K;       // points (center,sample pairs) per block
  constexpr int C0 = 32;           // 19 channels padded to 32
  __shared__ __align__(32) _Float16 sX[PB * C0];
  __shared__ __align__(32) _Float16 sY[PB * C1];
  __shared__ __align__(32) _Float16 sZ[PB * C2];
  const int b  = (blockIdx.x * CB) / NPOINT;
  const int s0 = (blockIdx.x * CB) % NPOINT;

  // stage 1: build x0 = concat(rel_xyz(3), features(16), zeros(13)) in LDS
  for (int e = threadIdx.x; e < PB * 2; e += blockDim.x) {
    int lp = e >> 1, hf = e & 1;
    int ci = lp / K;
    int s  = s0 + ci;
    int gi = gidx[((size_t)b * NPOINT + s) * K + (lp - ci * K)];
    const float* pc = new_xyz + ((size_t)b * NPOINT + s) * 3;
    const float* pg = xyz + ((size_t)b * NPTS + gi) * 3;
    const float* pf = feat + (size_t)b * NFEAT * NPTS + gi;
    _Float16* dst = sX + (size_t)lp * C0 + hf * 16;
    #pragma unroll
    for (int o = 0; o < 16; ++o) {
      int ch = hf * 16 + o;
      float v;
      if (ch < 3)       v = pg[ch] - pc[ch];
      else if (ch < 19) v = pf[(size_t)(ch - 3) * NPTS];
      else              v = 0.f;
      dst[o] = (_Float16)v;
    }
  }
  __syncthreads();
  mlp_layer<C0, C1, PB>(sX, sY, W1, b1);
  __syncthreads();
  mlp_layer<C1, C2, PB>(sY, sZ, W2, b2);
  __syncthreads();

  // stage 3: masked max over K (relu output >= 0, so 0-init matches x*mask semantics)
  for (int e = threadIdx.x; e < CB * C2; e += blockDim.x) {
    int ci = e / C2, ch = e - ci * C2;
    float m = 0.f;
    if (cnt[b * NPOINT + s0 + ci] > 0) {
      #pragma unroll 4
      for (int k = 0; k < K; ++k)
        m = fmaxf(m, (float)sZ[(size_t)(ci * K + k) * C2 + ch]);
    }
    aggin[((size_t)b * NPOINT + s0 + ci) * 192 + choff + ch] = (_Float16)m;
  }
}

// ---------------- final aggregation: 192 -> 128 WMMA, BN folded, relu, f32 out
__global__ __launch_bounds__(256) void agg_wmma_kernel(
    const _Float16* __restrict__ aggin, const _Float16* __restrict__ W,
    const float* __restrict__ bias, float* __restrict__ nf)
{
  const int lane   = threadIdx.x & 31;
  const int wave   = threadIdx.x >> 5;
  const int l15    = lane & 15;
  const int base8  = (lane < 16) ? 0 : 8;
  const int base16 = (lane < 16) ? 0 : 16;
  int tile = blockIdx.x * 8 + wave;       // 8 m-tiles x 512 n-tiles = 4096
  int mt = tile & 7;
  int nt = tile >> 3;
  v8f acc = {};
  const _Float16* wrow = W + (size_t)(mt * 16 + l15) * 192;
  const _Float16* arow = aggin + (size_t)(nt * 16 + l15) * 192;
  #pragma unroll
  for (int k0 = 0; k0 < 192; k0 += 32) {
    v16h af = load_frag_a(wrow, k0, base8);
    v16h bf = load_frag_b(arow, k0, base16);
    acc = __builtin_amdgcn_wmma_f32_16x16x32_f16(false, af, false, bf,
                                                 (short)0, acc, false, false);
  }
  int p  = nt * 16 + l15;        // flat point = b*NPOINT + s
  int bb = p >> 12;
  int s  = p & (NPOINT - 1);
  #pragma unroll
  for (int i = 0; i < 8; ++i) {
    int ch = mt * 16 + base8 + i;
    float v = acc[i] + bias[ch];
    nf[((size_t)bb * 128 + ch) * NPOINT + s] = (v > 0.f) ? v : 0.f;
  }
}

// ---------------- host launcher ----------------
extern "C" void kernel_launch(void* const* d_in, const int* in_sizes, int n_in,
                              void* d_out, int out_size, void* d_ws, size_t ws_size,
                              hipStream_t stream)
{
  (void)n_in; (void)out_size; (void)ws_size;
  struct L { const float *w, *g, *be, *me, *va; int cout, cin; } Ls[5];
  const float *xyz, *features, *density;
  const float** F = (const float**)d_in;

  // Two plausible flattenings of setup_inputs(): jax tree_flatten (dicts sorted
  // by key -> density first, in_sizes[0]==32768) vs insertion order
  // (xyz first, in_sizes[0]==98304). Detect via in_sizes[0].
  if (in_sizes[0] == B_N * 1 * NPTS) {
    // sorted pytree order: density, features, agg{bn{beta,gamma,mean,var}, w},
    // scales[0]{bns[0..1]{b,g,m,v}, ws[0..1]}, scales[1]{...}, xyz
    density  = F[0]; features = F[1]; xyz = F[27];
    Ls[0] = { F[15], F[8],  F[7],  F[9],  F[10],  32,  19 };
    Ls[1] = { F[16], F[12], F[11], F[13], F[14],  64,  32 };
    Ls[2] = { F[25], F[18], F[17], F[19], F[20],  64,  19 };
    Ls[3] = { F[26], F[22], F[21], F[23], F[24], 128,  64 };
    Ls[4] = { F[6],  F[3],  F[2],  F[4],  F[5],  128, 192 };
  } else {
    // insertion order: xyz, features, density, scales[0]{ws0,ws1,bn0{g,b,m,v},bn1},
    // scales[1]{...}, agg{w, bn{g,b,m,v}}
    xyz = F[0]; features = F[1]; density = F[2];
    Ls[0] = { F[3],  F[5],  F[6],  F[7],  F[8],   32,  19 };
    Ls[1] = { F[4],  F[9],  F[10], F[11], F[12],  64,  32 };
    Ls[2] = { F[13], F[15], F[16], F[17], F[18],  64,  19 };
    Ls[3] = { F[14], F[19], F[20], F[21], F[22], 128,  64 };
    Ls[4] = { F[23], F[24], F[25], F[26], F[27], 128, 192 };
  }

  // d_out: new_xyz (B,4096,3) | nf (B,128,4096) | new_density (B,1,4096), all f32
  float* new_xyz     = (float*)d_out;
  float* nf          = new_xyz + (size_t)B_N * NPOINT * 3;
  float* new_density = nf + (size_t)B_N * 128 * NPOINT;

  // workspace bump allocator (256B aligned)
  char* wp = (char*)d_ws;
  auto alloc = [&](size_t bytes) -> char* {
    char* r = wp; wp += (bytes + 255) & ~(size_t)255; return r;
  };
  int* gidx0 = (int*)alloc((size_t)B_N * NPOINT * 16 * sizeof(int));
  int* gidx1 = (int*)alloc((size_t)B_N * NPOINT * 32 * sizeof(int));
  int* cnt0  = (int*)alloc((size_t)B_N * NPOINT * sizeof(int));
  int* cnt1  = (int*)alloc((size_t)B_N * NPOINT * sizeof(int));
  const int cinpad[5] = { 32, 32, 32, 64, 192 };
  _Float16* W16[5]; float* bias[5];
  for (int l = 0; l < 5; ++l) {
    W16[l]  = (_Float16*)alloc((size_t)Ls[l].cout * cinpad[l] * sizeof(_Float16));
    bias[l] = (float*)alloc((size_t)Ls[l].cout * sizeof(float));
  }
  _Float16* aggin = (_Float16*)alloc((size_t)B_N * NPOINT * 192 * sizeof(_Float16));

  // 1) fold BN into f16 weights + f32 bias
  for (int l = 0; l < 5; ++l) {
    int total = Ls[l].cout * cinpad[l];
    fold_bn_kernel<<<(total + 255) / 256, 256, 0, stream>>>(
        Ls[l].w, Ls[l].g, Ls[l].be, Ls[l].me, Ls[l].va,
        Ls[l].cout, Ls[l].cin, cinpad[l], W16[l], bias[l]);
  }

  // 2) furthest point sampling (also emits new_xyz, new_density)
  fps_kernel<<<B_N, 1024, 0, stream>>>(xyz, density, new_xyz, new_density);

  // 3) ball query per scale (one wave per center; 8192 centers / 8 waves per block)
  ball_query_kernel<16><<<(B_N * NPOINT) / 8, 256, 0, stream>>>(
      xyz, new_xyz, gidx0, cnt0, 0.5f * 0.5f);
  ball_query_kernel<32><<<(B_N * NPOINT) / 8, 256, 0, stream>>>(
      xyz, new_xyz, gidx1, cnt1, 1.0f * 1.0f);

  // 4) fused group + MLP(WMMA) + masked max-pool per scale
  //    scale 0: K=16, 8 centers/block (128 pts), 19->32->64, channel offset 0
  fused_scale_kernel<16, 8, 32, 64><<<(B_N * NPOINT) / 8, 256, 0, stream>>>(
      xyz, features, new_xyz, gidx0, cnt0,
      W16[0], bias[0], W16[1], bias[1], aggin, 0);
  //    scale 1: K=32, 2 centers/block (64 pts), 19->64->128, channel offset 64
  fused_scale_kernel<32, 2, 64, 128><<<(B_N * NPOINT) / 2, 256, 0, stream>>>(
      xyz, features, new_xyz, gidx1, cnt1,
      W16[2], bias[2], W16[3], bias[3], aggin, 64);

  // 5) aggregation 192->128 (WMMA) + BN + relu -> nf
  agg_wmma_kernel<<<512, 256, 0, stream>>>(aggin, W16[4], bias[4], nf);
}
